// UncertaintyAwareVariationalBatchGAT_67413806678031
// MI455X (gfx1250) — matrix operand order repaired and checked by reading, not compiled, not run on hardware
//
#include <hip/hip_runtime.h>
#include <cstdint>

typedef __attribute__((ext_vector_type(16))) _Float16 v16h;
typedef __attribute__((ext_vector_type(8)))  _Float16 v8h;
typedef __attribute__((ext_vector_type(8)))  float    v8f;

// 16-bit A-matrix (16x32) K index for (lane, vgpr-pair v), element 0 of the pair: contiguous k = 16*(lane/16) + 2v.
__device__ __forceinline__ int kpatA(int lane, int v) {
  int base = ((lane >> 4) & 1) * 16;
  return base + 2 * v;
}

__global__ __launch_bounds__(256)
void gat_mc_kernel(const float* __restrict__ x, const float* __restrict__ emb,
                   const int* __restrict__ adj,
                   const float* __restrict__ w1, const float* __restrict__ as1,
                   const float* __restrict__ ad1, const float* __restrict__ b1,
                   const float* __restrict__ w2, const float* __restrict__ as2,
                   const float* __restrict__ ad2, const float* __restrict__ b2,
                   float* __restrict__ wsf) {
  const int t    = threadIdx.x;
  const int lane = t & 31;
  const int wv   = t >> 5;
  const int blk  = blockIdx.x;
  const int s    = blk >> 7;   // MC sample 0..3
  const int b    = blk & 127;  // batch

  __shared__ uint32_t smask[128 * 4];      // adjacency bitmask (2 KB)
  __shared__ _Float16 sw1[128 * 128];      // w1[s] f16, B-fragment layout (32 KB)
  __shared__ _Float16 shAT[128 * 128];     // h transposed: [(h*16+o)][n]    (32 KB)
  __shared__ _Float16 sxs2[128 * 128];     // concat-head layer-2 input      (32 KB)
  __shared__ float    sh2[128 * 2];
  __shared__ float    ssrc[128], sdst[128], srmax[128], srscale[128];
  __shared__ float    sas1[128], sad1[128], sb1v[16];
  __shared__ float    sw2[256], sas2v[2], sad2v[2], sb2v[2];

  // ---- Phase A: adj bitmask (ballot), w1 -> LDS f16 in B-frag layout, small weights
  {
    const int* abase = adj + (size_t)b * 128 * 128;
    for (int it = 0; it < 64; ++it) {
      int idx = it * 256 + t;
      unsigned long long bal = __ballot(abase[idx] != 0);
      if (lane == 0) smask[it * 8 + wv] = (uint32_t)bal;
    }
  }
  {
    const float* w1s = w1 + (size_t)s * 16384;
    for (int i = 0; i < 64; ++i) {
      int g = i * 256 + t;   // g = h*2048 + k*16 + o
      // B-fragment layout = nibble swap: [h | k>>4 | o | k&15]
      int dst = (g & ~0xFF) | ((g & 0xF) << 4) | ((g >> 4) & 0xF);
      sw1[dst] = (_Float16)w1s[g];
    }
  }
  if (t < 128) sas1[t] = as1[s * 128 + t];
  else         sad1[t - 128] = ad1[s * 128 + (t - 128)];
  if (t < 16)  sb1v[t] = b1[t];
  sw2[t] = w2[s * 256 + t];
  if (t < 2) { sas2v[t] = as2[s * 2 + t]; sad2v[t] = ad2[s * 2 + t]; sb2v[t] = b2[t]; }
  __syncthreads();

  const int mrow = wv * 16 + (lane & 15);   // A-matrix row handled by this lane
  const int col  = lane & 15;               // B/D column handled by this lane
  const int lhi  = (lane >> 4) & 1;
  const float2* x2 = (const float2*)(x   + ((size_t)b * 128 + mrow) * 64);
  const float2* e2 = (const float2*)(emb + ((size_t)b * 128 + mrow) * 64);

  // ---- Phase B1: all-head GEMM1:  H = x0 @ W1   (128x128)@(128x128)
  {
    v8f c[8];
    #pragma unroll
    for (int h = 0; h < 8; ++h) c[h] = (v8f){0.f, 0.f, 0.f, 0.f, 0.f, 0.f, 0.f, 0.f};
    for (int kt = 0; kt < 4; ++kt) {
      v16h af;
      #pragma unroll
      for (int v = 0; v < 8; ++v) {
        int ka = kt * 32 + kpatA(lane, v);         // even
        float2 p = (ka < 64) ? x2[ka >> 1] : e2[(ka - 64) >> 1];
        af[2 * v]     = (_Float16)p.x;
        af[2 * v + 1] = (_Float16)p.y;
      }
      #pragma unroll
      for (int h = 0; h < 8; ++h) {
        v16h bf = *(const v16h*)&sw1[h * 2048 + (kt * 2 + lhi) * 256 + col * 16];
        c[h] = __builtin_amdgcn_wmma_f32_16x16x32_f16(false, af, false, bf,
                                                      (short)0, c[h], false, false);
      }
    }
    int rbase = wv * 16 + ((lane < 16) ? 0 : 8);
    #pragma unroll
    for (int h = 0; h < 8; ++h) {
      v8h pk;
      #pragma unroll
      for (int r = 0; r < 8; ++r) pk[r] = (_Float16)c[h][r];
      *(v8h*)&shAT[(h * 16 + col) * 128 + rbase] = pk;   // contiguous 16B store
    }
  }
  __syncthreads();

  // ---- Phase B2: per-head attention
  for (int h = 0; h < 8; ++h) {
    // attention source / destination scores: dot(tanh(h_n), a)
    {
      int n = t & 127;
      const float* av = (t < 128) ? &sas1[h * 16] : &sad1[h * 16];
      float acc = 0.f;
      #pragma unroll
      for (int o = 0; o < 16; ++o)
        acc += tanhf((float)shAT[(h * 16 + o) * 128 + n]) * av[o];
      if (t < 128) ssrc[n] = acc; else sdst[n] = acc;
    }
    __syncthreads();

    // per-row softmax stats (max, 1/sum), 2 threads per row
    {
      int i = t >> 1, half = t & 1;
      float si = ssrc[i];
      float mx = -1e30f;
      for (int jj = 0; jj < 64; ++jj) {
        int j = half * 64 + jj;
        uint32_t bit = (smask[i * 4 + (j >> 5)] >> (j & 31)) & 1u;
        float v = si + sdst[j];
        float lg = v > 0.f ? v : 0.2f * v;
        mx = bit ? fmaxf(mx, lg) : mx;
      }
      mx = fmaxf(mx, __shfl_xor(mx, 1, 32));
      float sum = 0.f;
      for (int jj = 0; jj < 64; ++jj) {
        int j = half * 64 + jj;
        uint32_t bit = (smask[i * 4 + (j >> 5)] >> (j & 31)) & 1u;
        float v = si + sdst[j];
        float lg = v > 0.f ? v : 0.2f * v;
        sum += bit ? __expf(lg - mx) : 0.f;
      }
      sum += __shfl_xor(sum, 1, 32);
      if (!half) { srmax[i] = mx; srscale[i] = 1.f / sum; }
    }
    __syncthreads();

    // attention GEMM: out = softmax(E) @ h ; A built on the fly, B one wide LDS load
    {
      v8f d = {0.f, 0.f, 0.f, 0.f, 0.f, 0.f, 0.f, 0.f};
      float si = ssrc[mrow];
      float mi = srmax[mrow];
      const uint32_t* mrk = &smask[mrow * 4];
      const _Float16* bcol = &shAT[(h * 16 + col) * 128];
      for (int kt = 0; kt < 4; ++kt) {
        v16h af;
        #pragma unroll
        for (int v = 0; v < 8; ++v) {
          int ka = kt * 32 + kpatA(lane, v);
          #pragma unroll
          for (int e = 0; e < 2; ++e) {
            int j = ka + e;
            uint32_t bit = (mrk[j >> 5] >> (j & 31)) & 1u;
            float vv = si + sdst[j];
            float lg = vv > 0.f ? vv : 0.2f * vv;
            af[2 * v + e] = (_Float16)(bit ? __expf(lg - mi) : 0.f);
          }
        }
        v16h bf = *(const v16h*)&bcol[kt * 32 + lhi * 16];
        d = __builtin_amdgcn_wmma_f32_16x16x32_f16(false, af, false, bf,
                                                   (short)0, d, false, false);
      }
      #pragma unroll
      for (int r = 0; r < 8; ++r) {
        int row = wv * 16 + ((lane < 16) ? r : r + 8);
        float val = d[r] * srscale[row] + sb1v[col];
        val = val > 0.f ? val : (__expf(val) - 1.f);      // ELU
        sxs2[row * 128 + h * 16 + col] = (_Float16)val;   // concat heads
      }
    }
    __syncthreads();
  }

  // ---- Layer 2 GEMM: h2 = xs2 @ w2[s,0]  (128x128)@(128x2), cols 2..15 zero
  {
    v8f d = {0.f, 0.f, 0.f, 0.f, 0.f, 0.f, 0.f, 0.f};
    const _Float16* arow = &sxs2[mrow * 128];
    for (int kt = 0; kt < 4; ++kt) {
      v16h af, bf;
      #pragma unroll
      for (int v = 0; v < 8; ++v) {
        int ka = kt * 32 + kpatA(lane, v);
        af[2 * v]     = arow[ka];
        af[2 * v + 1] = arow[ka + 1];
        int kb = kt * 32 + lhi * 16 + 2 * v;
        _Float16 z = (_Float16)0.f;
        bf[2 * v]     = (col < 2) ? (_Float16)sw2[kb * 2 + col] : z;
        bf[2 * v + 1] = (col < 2) ? (_Float16)sw2[(kb + 1) * 2 + col] : z;
      }
      d = __builtin_amdgcn_wmma_f32_16x16x32_f16(false, af, false, bf,
                                                 (short)0, d, false, false);
    }
    if (col < 2) {
      #pragma unroll
      for (int r = 0; r < 8; ++r) {
        int row = wv * 16 + ((lane < 16) ? r : r + 8);
        sh2[row * 2 + col] = d[r];
      }
    }
  }
  __syncthreads();

  // ---- layer-2 attention scores
  if (t < 128) {
    float t0 = tanhf(sh2[t * 2 + 0]);
    float t1 = tanhf(sh2[t * 2 + 1]);
    ssrc[t] = t0 * sas2v[0] + t1 * sas2v[1];
    sdst[t] = t0 * sad2v[0] + t1 * sad2v[1];
  }
  __syncthreads();

  // ---- only the ego node (row 127) matters: softmax row, out, log_softmax
  if (t == 0) {
    float si = ssrc[127];
    const uint32_t* mrk = &smask[127 * 4];
    float mx = -1e30f;
    for (int j = 0; j < 128; ++j) {
      uint32_t bit = (mrk[j >> 5] >> (j & 31)) & 1u;
      float v = si + sdst[j];
      float lg = v > 0.f ? v : 0.2f * v;
      if (bit) mx = fmaxf(mx, lg);
    }
    float sum = 0.f, o0 = 0.f, o1 = 0.f;
    for (int j = 0; j < 128; ++j) {
      uint32_t bit = (mrk[j >> 5] >> (j & 31)) & 1u;
      float v = si + sdst[j];
      float lg = v > 0.f ? v : 0.2f * v;
      float e = bit ? __expf(lg - mx) : 0.f;
      sum += e;
      o0 += e * sh2[j * 2 + 0];
      o1 += e * sh2[j * 2 + 1];
    }
    o0 = o0 / sum + sb2v[0];
    o1 = o1 / sum + sb2v[1];
    float mo  = fmaxf(o0, o1);
    float lse = mo + __logf(__expf(o0 - mo) + __expf(o1 - mo));
    wsf[((size_t)s * 128 + b) * 2 + 0] = o0 - lse;
    wsf[((size_t)s * 128 + b) * 2 + 1] = o1 - lse;
  }
}

// Deterministic mean over the S=4 MC samples.
__global__ __launch_bounds__(256)
void gat_reduce_kernel(const float* __restrict__ wsf, float* __restrict__ out) {
  int t = threadIdx.x;  // 256 = b*2 outputs
  float acc = 0.f;
  #pragma unroll
  for (int s = 0; s < 4; ++s) acc += wsf[s * 256 + t];
  out[t] = acc * 0.25f;
}

extern "C" void kernel_launch(void* const* d_in, const int* in_sizes, int n_in,
                              void* d_out, int out_size, void* d_ws, size_t ws_size,
                              hipStream_t stream) {
  (void)in_sizes; (void)n_in; (void)out_size; (void)ws_size;
  const float* x   = (const float*)d_in[0];
  const float* emb = (const float*)d_in[1];
  const int*   adj = (const int*)d_in[2];
  const float* w1  = (const float*)d_in[3];
  const float* as1 = (const float*)d_in[4];
  const float* ad1 = (const float*)d_in[5];
  const float* b1  = (const float*)d_in[6];
  const float* w2  = (const float*)d_in[7];
  const float* as2 = (const float*)d_in[8];
  const float* ad2 = (const float*)d_in[9];
  const float* b2  = (const float*)d_in[10];
  float* wsf = (float*)d_ws;

  gat_mc_kernel<<<512, 256, 0, stream>>>(x, emb, adj, w1, as1, ad1, b1,
                                         w2, as2, ad2, b2, wsf);
  gat_reduce_kernel<<<1, 256, 0, stream>>>(wsf, (float*)d_out);
}
